// SimpleCrossAttentionTransformerBlock_10247791968530
// MI455X (gfx1250) — matrix operand order repaired and checked by reading, not compiled, not run on hardware
//
#include <hip/hip_runtime.h>
#include <stdint.h>

// Problem constants (from reference)
#define D_   1024
#define B_   2
#define N_   2048          // Nq == Nk
#define H_   16
#define HD_  64
#define DFF_ 4096
#define BN_  (B_ * N_)     // 4096 token rows

typedef __bf16 bf16_t;
typedef __attribute__((ext_vector_type(16))) __bf16 v16bf;
typedef __attribute__((ext_vector_type(8)))  float  v8f;
typedef __attribute__((ext_vector_type(4)))  int    v4i;

union BF16x16 { v16bf v; uint4 q[2]; };

#define USE_ASYNC_LDS __has_builtin(__builtin_amdgcn_global_load_async_to_lds_b128)

#if USE_ASYNC_LDS
#if __has_builtin(__builtin_amdgcn_s_wait_asynccnt)
#define WAIT_ASYNC(n) __builtin_amdgcn_s_wait_asynccnt(n)
#else
#define WAIT_ASYNC(n) asm volatile("s_wait_asynccnt " #n ::: "memory")
#endif
#else
#define WAIT_ASYNC(n)
#endif

// Pointer types the async builtin expects (b128 payload = int4 pointee):
typedef __attribute__((address_space(1))) v4i* gptr_v4i;   // global
typedef __attribute__((address_space(3))) v4i* lptr_v4i;   // LDS

// 16-byte global -> LDS copy (per lane). Async DMA path on CDNA5, tracked by
// ASYNCcnt; falls back to load+ds_store if the builtin is unavailable.
__device__ __forceinline__ void async_cp16(const bf16_t* g, bf16_t* l) {
#if USE_ASYNC_LDS
  __builtin_amdgcn_global_load_async_to_lds_b128((gptr_v4i)(g), (lptr_v4i)(l), 0, 0);
#else
  *(uint4*)l = *(const uint4*)g;
#endif
}

__device__ __forceinline__ bf16_t f2bf(float f) {
  union { float f; uint32_t u; } c; c.f = f;
  uint32_t u = c.u + 0x7FFFu + ((c.u >> 16) & 1u);   // round-to-nearest-even
  union { uint16_t s; bf16_t b; } r; r.s = (uint16_t)(u >> 16);
  return r.b;
}

__device__ __forceinline__ v8f wmma_bf16(v16bf a, v16bf b, v8f c) {
  // D = A(16x32 bf16) * B(32x16 bf16) + C(16x16 f32)
  return __builtin_amdgcn_wmma_f32_16x16x32_bf16(false, a, false, b, (short)0, c,
                                                 false, false);
}

// ---------------------------------------------------------------------------
// fp32 -> bf16 elementwise conversion (weights). 4 elems/thread.
// ---------------------------------------------------------------------------
__global__ void __launch_bounds__(256) cvt_bf16_kernel(const float* __restrict__ in,
                                                       bf16_t* __restrict__ out,
                                                       int n4) {
  int i = blockIdx.x * 256 + threadIdx.x;
  if (i < n4) {
    float4 v = ((const float4*)in)[i];
    union { bf16_t e[4]; uint2 u; } o;
    o.e[0] = f2bf(v.x); o.e[1] = f2bf(v.y); o.e[2] = f2bf(v.z); o.e[3] = f2bf(v.w);
    ((uint2*)out)[i] = o.u;
  }
}

// ---------------------------------------------------------------------------
// LayerNorm over D=1024, one block (256 thr) per row; bf16 output.
// ---------------------------------------------------------------------------
__global__ void __launch_bounds__(256) ln_bf16_kernel(const float* __restrict__ x,
                                                      const float* __restrict__ g,
                                                      const float* __restrict__ bta,
                                                      bf16_t* __restrict__ out) {
  const int row = blockIdx.x;
  const int t = threadIdx.x;
  __shared__ float red[8];

  float4 v = ((const float4*)(x + (size_t)row * D_))[t];
  float s = v.x + v.y + v.z + v.w;
  for (int o = 16; o > 0; o >>= 1) s += __shfl_down(s, o, 32);
  if ((t & 31) == 0) red[t >> 5] = s;
  __syncthreads();
  s = red[0] + red[1] + red[2] + red[3] + red[4] + red[5] + red[6] + red[7];
  const float mu = s * (1.0f / D_);

  float dx = v.x - mu, dy = v.y - mu, dz = v.z - mu, dw = v.w - mu;
  float s2 = dx * dx + dy * dy + dz * dz + dw * dw;
  __syncthreads();
  for (int o = 16; o > 0; o >>= 1) s2 += __shfl_down(s2, o, 32);
  if ((t & 31) == 0) red[t >> 5] = s2;
  __syncthreads();
  s2 = red[0] + red[1] + red[2] + red[3] + red[4] + red[5] + red[6] + red[7];
  const float rstd = rsqrtf(s2 * (1.0f / D_) + 1e-5f);

  float4 gv = ((const float4*)g)[t];
  float4 bv = ((const float4*)bta)[t];
  union { bf16_t e[4]; uint2 u; } o;
  o.e[0] = f2bf(dx * rstd * gv.x + bv.x);
  o.e[1] = f2bf(dy * rstd * gv.y + bv.y);
  o.e[2] = f2bf(dz * rstd * gv.z + bv.z);
  o.e[3] = f2bf(dw * rstd * gv.w + bv.w);
  ((uint2*)(out + (size_t)row * D_))[t] = o.u;
}

// ---------------------------------------------------------------------------
// WMMA bf16 GEMM: C[M,N] = X[M,K] * W[N,K]^T  (f32 accumulate)
// Block = 256 thr = 8 waves (2 M-waves x 4 N-waves); wave tile = 32x64,
// block tile = 64x256. A/B K-slices (k=32) are staged into double-buffered
// LDS via async global->LDS DMA (ASYNCcnt) and consumed with ds_load_b128.
// Rows padded to 40 elems (80B = 20 banks) => conflict-free frag reads.
// Epilogue MODE: 0=Q-proj scatter, 1=KV-proj scatter (K + V-transposed),
//                2=Wo (+bias+resid -> f32 & bf16), 3=FFN1 (+bias, SiLU -> bf16),
//                4=FFN2 (+bias+resid -> f32 d_out)
// ---------------------------------------------------------------------------
#define AP 40   // padded LDS row pitch in elements for 32-wide K slices

template <int MODE>
__global__ void __launch_bounds__(256) gemm_bf16(const bf16_t* __restrict__ X,
                                                 const bf16_t* __restrict__ W,
                                                 int N, int K,
                                                 const float* __restrict__ bias,
                                                 const float* __restrict__ resid,
                                                 float* __restrict__ outf,
                                                 bf16_t* __restrict__ pA,
                                                 bf16_t* __restrict__ pB) {
  __shared__ bf16_t Abuf[2][64 * AP];     // 10.2 KB
  __shared__ bf16_t Bbuf[2][256 * AP];    // 41.0 KB

  const int t = threadIdx.x;
  const int lane = t & 31;
  const int w = t >> 5;
  const int wm = w & 1, wn = w >> 1;
  const int hl = lane >> 4;               // lane half (0/1)
  const int lr = lane & 15;
  const int bm0 = blockIdx.y * 64;
  const int bn0 = blockIdx.x * 256;
  const int m0l = wm * 32;                // wave tile origin inside block tile
  const int n0l = wn * 64;

  // staging map: each thread copies one 16B chunk of A and four of B per stage
  const int crow = t >> 2;                // 0..63
  const int cchk = (t & 3) * 8;           // element offset 0,8,16,24

  const bf16_t* Xg  = X + (size_t)(bm0 + crow) * K + cchk;
  const bf16_t* Wg0 = W + (size_t)(bn0 + crow) * K + cchk;

  auto issue = [&](int k0, int buf) {
    async_cp16(Xg + k0, &Abuf[buf][crow * AP + cchk]);
#pragma unroll
    for (int i = 0; i < 4; i++)
      async_cp16(Wg0 + (size_t)i * 64 * K + k0,
                 &Bbuf[buf][(crow + i * 64) * AP + cchk]);
  };

  v8f acc[2][4];
  const v8f zf = {0.f, 0.f, 0.f, 0.f, 0.f, 0.f, 0.f, 0.f};
#pragma unroll
  for (int i = 0; i < 2; i++)
#pragma unroll
    for (int j = 0; j < 4; j++) acc[i][j] = zf;

  const int nk = K >> 5;
  issue(0, 0);
  for (int ks = 0; ks < nk; ks++) {
    const int cur = ks & 1;
    if (ks + 1 < nk) {
      issue((ks + 1) << 5, cur ^ 1);
      WAIT_ASYNC(5);                      // stage ks fully landed (in-order)
    } else {
      WAIT_ASYNC(0);
    }
    __syncthreads();                      // publish buf[cur] to all waves

    const bf16_t* Al = Abuf[cur];
    const bf16_t* Bl = Bbuf[cur];
    BF16x16 a[2], b[4];
#pragma unroll
    for (int i = 0; i < 2; i++) {
      const bf16_t* p = Al + (size_t)(m0l + i * 16 + lr) * AP + hl * 8;
      a[i].q[0] = *(const uint4*)(p);     // K = hl*8 .. +7
      a[i].q[1] = *(const uint4*)(p + 16);// K = 16+hl*8 .. +7
    }
#pragma unroll
    for (int j = 0; j < 4; j++) {
      const bf16_t* p = Bl + (size_t)(n0l + j * 16 + lr) * AP + hl * 8;
      b[j].q[0] = *(const uint4*)(p);     // K = hl*8 .. +7
      b[j].q[1] = *(const uint4*)(p + 16);// K = 16+hl*8 .. +7
    }
#pragma unroll
    for (int i = 0; i < 2; i++)
#pragma unroll
      for (int j = 0; j < 4; j++) acc[i][j] = wmma_bf16(a[i].v, b[j].v, acc[i][j]);

    __syncthreads();                      // everyone done reading buf[cur]
  }

  // Epilogue. C layout: VGPR jj holds row m = jj + 8*hl, col n = lr (per tile).
#pragma unroll
  for (int i = 0; i < 2; i++) {
#pragma unroll
    for (int j = 0; j < 4; j++) {
      const int n = bn0 + n0l + j * 16 + lr;
      const int mb = bm0 + m0l + i * 16 + 8 * hl;
#pragma unroll
      for (int jj = 0; jj < 8; jj++) {
        const int m = mb + jj;
        float v = acc[i][j][jj];
        if (MODE == 0) {                       // Q: [B,H,N,64] bf16
          int bb = m >> 11, tok = m & (N_ - 1);
          int hh = n >> 6, d = n & 63;
          pA[(((size_t)bb * H_ + hh) * N_ + tok) * HD_ + d] = f2bf(v);
        } else if (MODE == 1) {                // K: [B,H,N,64]; V^T: [B,H,64,N]
          int bb = m >> 11, tok = m & (N_ - 1);
          if (n < D_) {
            int hh = n >> 6, d = n & 63;
            pA[(((size_t)bb * H_ + hh) * N_ + tok) * HD_ + d] = f2bf(v);
          } else {
            int nn = n - D_;
            int hh = nn >> 6, d = nn & 63;
            pB[(((size_t)bb * H_ + hh) * HD_ + d) * N_ + tok] = f2bf(v);
          }
        } else if (MODE == 2) {                // Wo: + bo + q_x -> x (f32 & bf16)
          v += bias[n] + resid[(size_t)m * D_ + n];
          outf[(size_t)m * D_ + n] = v;
          pA[(size_t)m * D_ + n] = f2bf(v);
        } else if (MODE == 3) {                // FFN1: SiLU(x@W1^T + b1) -> bf16
          v += bias[n];
          float sv = v / (1.0f + __expf(-v));
          pA[(size_t)m * DFF_ + n] = f2bf(sv);
        } else {                               // FFN2: + b2 + x -> d_out f32
          v += bias[n] + resid[(size_t)m * D_ + n];
          outf[(size_t)m * D_ + n] = v;
        }
      }
    }
  }
}

// ---------------------------------------------------------------------------
// Flash attention: one wave per (b,h, 16-query tile). HD=64, 64 keys/iter.
// Q [B,H,N,64] bf16, K [B,H,N,64] bf16, V^T [B,H,64,N] bf16 -> attn [BN,D] bf16
// ---------------------------------------------------------------------------
__global__ void __launch_bounds__(32) attn_kernel(const bf16_t* __restrict__ Qh,
                                                  const bf16_t* __restrict__ Kh,
                                                  const bf16_t* __restrict__ VTh,
                                                  bf16_t* __restrict__ attnb) {
  const int qt = blockIdx.x & (N_ / 16 - 1);   // query tile (0..127)
  const int bh = blockIdx.x >> 7;              // 0..31
  const int bb = bh >> 4, hh = bh & 15;
  const int lane = threadIdx.x;
  const int hl = lane >> 4, lr = lane & 15;

  const bf16_t* Qbase = Qh + ((size_t)bh * N_ + qt * 16) * HD_;
  const bf16_t* Kbase = Kh + (size_t)bh * N_ * HD_;
  const bf16_t* Vbase = VTh + (size_t)bh * HD_ * N_;

  __shared__ bf16_t Pst[16 * 64];              // 2 KB P-staging tile

  // Q A-frags (reused for all 2048 keys): rows = lr, k-splits 0..31 / 32..63
  BF16x16 q[2];
#pragma unroll
  for (int i = 0; i < 2; i++) {
    const bf16_t* p = Qbase + (size_t)lr * HD_ + i * 32 + hl * 8;
    q[i].q[0] = *(const uint4*)(p);
    q[i].q[1] = *(const uint4*)(p + 16);
  }

  const v8f zf = {0.f, 0.f, 0.f, 0.f, 0.f, 0.f, 0.f, 0.f};
  v8f O[4]; for (int d = 0; d < 4; d++) O[d] = zf;
  float rowm[8], rowl[8];
#pragma unroll
  for (int j = 0; j < 8; j++) { rowm[j] = -3.0e38f; rowl[j] = 0.0f; }

  const float scale = 0.125f;                  // HD^-0.5

  for (int kt = 0; kt < N_; kt += 64) {
    // ---- S = (Q K^T) * scale : 4 key tiles of 16
    v8f S[4];
#pragma unroll
    for (int nt = 0; nt < 4; nt++) {
      const bf16_t* kr = Kbase + (size_t)(kt + nt * 16 + lr) * HD_;
      BF16x16 kb0, kb1;
      kb0.q[0] = *(const uint4*)(kr + hl * 8);
      kb0.q[1] = *(const uint4*)(kr + hl * 8 + 16);
      kb1.q[0] = *(const uint4*)(kr + 32 + hl * 8);
      kb1.q[1] = *(const uint4*)(kr + 32 + hl * 8 + 16);
      v8f s = zf;
      s = wmma_bf16(q[0].v, kb0.v, s);
      s = wmma_bf16(q[1].v, kb1.v, s);
#pragma unroll
      for (int jj = 0; jj < 8; jj++) s[jj] *= scale;
      S[nt] = s;
    }
    // ---- online softmax: row max across 4 tiles + 16 lanes
    float newm[8], alpha[8];
#pragma unroll
    for (int jj = 0; jj < 8; jj++) {
      float tv = fmaxf(fmaxf(S[0][jj], S[1][jj]), fmaxf(S[2][jj], S[3][jj]));
      tv = fmaxf(tv, __shfl_xor(tv, 1, 16));
      tv = fmaxf(tv, __shfl_xor(tv, 2, 16));
      tv = fmaxf(tv, __shfl_xor(tv, 4, 16));
      tv = fmaxf(tv, __shfl_xor(tv, 8, 16));
      newm[jj] = fmaxf(rowm[jj], tv);
      alpha[jj] = __expf(rowm[jj] - newm[jj]);
      rowm[jj] = newm[jj];
    }
    float rs[8];
#pragma unroll
    for (int jj = 0; jj < 8; jj++) rs[jj] = 0.0f;
#pragma unroll
    for (int nt = 0; nt < 4; nt++)
#pragma unroll
      for (int jj = 0; jj < 8; jj++) {
        float p = __expf(S[nt][jj] - newm[jj]);
        S[nt][jj] = p;
        rs[jj] += p;
      }
#pragma unroll
    for (int jj = 0; jj < 8; jj++) {
      float tv = rs[jj];
      tv += __shfl_xor(tv, 1, 16);
      tv += __shfl_xor(tv, 2, 16);
      tv += __shfl_xor(tv, 4, 16);
      tv += __shfl_xor(tv, 8, 16);
      rowl[jj] = rowl[jj] * alpha[jj] + tv;
    }
#pragma unroll
    for (int d = 0; d < 4; d++)
#pragma unroll
      for (int jj = 0; jj < 8; jj++) O[d][jj] *= alpha[jj];

    // ---- stage P (C-layout) -> LDS -> A-layout frags
#pragma unroll
    for (int nt = 0; nt < 4; nt++)
#pragma unroll
      for (int jj = 0; jj < 8; jj++)
        Pst[(jj + 8 * hl) * 64 + nt * 16 + lr] = f2bf(S[nt][jj]);
    asm volatile("s_wait_dscnt 0" ::: "memory");   // CDNA5 split-counter wait
    BF16x16 pf[2];
#pragma unroll
    for (int i = 0; i < 2; i++) {
      const bf16_t* p = Pst + lr * 64 + i * 32 + hl * 8;
      pf[i].q[0] = *(const uint4*)(p);
      pf[i].q[1] = *(const uint4*)(p + 16);
    }
    // ---- O += P V : V^T rows give contiguous B-frags
#pragma unroll
    for (int dt = 0; dt < 4; dt++) {
      const bf16_t* vr = Vbase + (size_t)(dt * 16 + lr) * N_ + kt;
      BF16x16 vb0, vb1;
      vb0.q[0] = *(const uint4*)(vr + hl * 8);
      vb0.q[1] = *(const uint4*)(vr + hl * 8 + 16);
      vb1.q[0] = *(const uint4*)(vr + 32 + hl * 8);
      vb1.q[1] = *(const uint4*)(vr + 32 + hl * 8 + 16);
      O[dt] = wmma_bf16(pf[0].v, vb0.v, O[dt]);
      O[dt] = wmma_bf16(pf[1].v, vb1.v, O[dt]);
    }
  }

  // ---- finalize: O /= l, scatter to attn [B*N, D] bf16
  float inv[8];
#pragma unroll
  for (int jj = 0; jj < 8; jj++) inv[jj] = 1.0f / rowl[jj];
#pragma unroll
  for (int dt = 0; dt < 4; dt++)
#pragma unroll
    for (int jj = 0; jj < 8; jj++) {
      int tok = qt * 16 + jj + 8 * hl;
      attnb[((size_t)bb * N_ + tok) * D_ + hh * HD_ + dt * 16 + lr] =
          f2bf(O[dt][jj] * inv[jj]);
    }
}

// ---------------------------------------------------------------------------
extern "C" void kernel_launch(void* const* d_in, const int* in_sizes, int n_in,
                              void* d_out, int out_size, void* d_ws, size_t ws_size,
                              hipStream_t stream) {
  (void)in_sizes; (void)n_in; (void)out_size; (void)ws_size;
  const float* q_x  = (const float*)d_in[0];
  const float* kv_x = (const float*)d_in[1];
  const float* Wq   = (const float*)d_in[2];
  const float* Wkv  = (const float*)d_in[3];
  const float* Wo   = (const float*)d_in[4];
  const float* bo   = (const float*)d_in[5];
  const float* lnqg = (const float*)d_in[6];
  const float* lnqb = (const float*)d_in[7];
  const float* lnkg = (const float*)d_in[8];
  const float* lnkb = (const float*)d_in[9];
  const float* W1   = (const float*)d_in[10];
  const float* b1   = (const float*)d_in[11];
  const float* W2   = (const float*)d_in[12];
  const float* b2   = (const float*)d_in[13];
  float* out = (float*)d_out;

  char* ws = (char*)d_ws;
  auto alloc = [&](size_t bytes) {
    char* p = ws; ws += (bytes + 255) & ~(size_t)255; return p;
  };
  bf16_t* Wq_b  = (bf16_t*)alloc((size_t)D_ * D_ * 2);
  bf16_t* Wkv_b = (bf16_t*)alloc((size_t)2 * D_ * D_ * 2);
  bf16_t* Wo_b  = (bf16_t*)alloc((size_t)D_ * D_ * 2);
  bf16_t* W1_b  = (bf16_t*)alloc((size_t)DFF_ * D_ * 2);
  bf16_t* W2_b  = (bf16_t*)alloc((size_t)D_ * DFF_ * 2);
  bf16_t* qln   = (bf16_t*)alloc((size_t)BN_ * D_ * 2);
  bf16_t* kvln  = (bf16_t*)alloc((size_t)BN_ * D_ * 2);
  bf16_t* Qh    = (bf16_t*)alloc((size_t)B_ * H_ * N_ * HD_ * 2);
  bf16_t* Kh    = (bf16_t*)alloc((size_t)B_ * H_ * N_ * HD_ * 2);
  bf16_t* VTh   = (bf16_t*)alloc((size_t)B_ * H_ * HD_ * N_ * 2);
  bf16_t* attnb = (bf16_t*)alloc((size_t)BN_ * D_ * 2);
  float*  x_f   = (float*)alloc((size_t)BN_ * D_ * 4);
  bf16_t* x_b   = (bf16_t*)alloc((size_t)BN_ * D_ * 2);
  bf16_t* h_b   = (bf16_t*)alloc((size_t)BN_ * DFF_ * 2);

  // 1) weights -> bf16
  cvt_bf16_kernel<<<(D_ * D_) / 1024, 256, 0, stream>>>(Wq, Wq_b, (D_ * D_) / 4);
  cvt_bf16_kernel<<<(2 * D_ * D_) / 1024, 256, 0, stream>>>(Wkv, Wkv_b, (2 * D_ * D_) / 4);
  cvt_bf16_kernel<<<(D_ * D_) / 1024, 256, 0, stream>>>(Wo, Wo_b, (D_ * D_) / 4);
  cvt_bf16_kernel<<<(DFF_ * D_) / 1024, 256, 0, stream>>>(W1, W1_b, (DFF_ * D_) / 4);
  cvt_bf16_kernel<<<(D_ * DFF_) / 1024, 256, 0, stream>>>(W2, W2_b, (D_ * DFF_) / 4);

  // 2) layernorms -> bf16
  ln_bf16_kernel<<<BN_, 256, 0, stream>>>(q_x, lnqg, lnqb, qln);
  ln_bf16_kernel<<<BN_, 256, 0, stream>>>(kv_x, lnkg, lnkb, kvln);

  // 3) projections (WMMA, async-LDS staged)
  gemm_bf16<0><<<dim3(D_ / 256, BN_ / 64), 256, 0, stream>>>(
      qln, Wq_b, D_, D_, nullptr, nullptr, nullptr, Qh, nullptr);
  gemm_bf16<1><<<dim3(2 * D_ / 256, BN_ / 64), 256, 0, stream>>>(
      kvln, Wkv_b, 2 * D_, D_, nullptr, nullptr, nullptr, Kh, VTh);

  // 4) flash attention (WMMA)
  attn_kernel<<<B_ * H_ * (N_ / 16), 32, 0, stream>>>(Qh, Kh, VTh, attnb);

  // 5) output projection + residual
  gemm_bf16<2><<<dim3(D_ / 256, BN_ / 64), 256, 0, stream>>>(
      attnb, Wo_b, D_, D_, bo, q_x, x_f, x_b, nullptr);

  // 6) FFN
  gemm_bf16<3><<<dim3(DFF_ / 256, BN_ / 64), 256, 0, stream>>>(
      x_b, W1_b, DFF_, D_, b1, nullptr, nullptr, h_b, nullptr);
  gemm_bf16<4><<<dim3(D_ / 256, BN_ / 64), 256, 0, stream>>>(
      h_b, W2_b, D_, DFF_, b2, x_f, out, nullptr, nullptr);
}
